// SWABlock_5866925326841
// MI455X (gfx1250) — matrix operand order repaired and checked by reading, not compiled, not run on hardware
//
#include <hip/hip_runtime.h>
#include <hip/hip_bf16.h>
#include <cstdint>
#include <cstddef>

// ---------------------------------------------------------------------------
// MI455X (gfx1250) sliding-window-attention transformer block.
// bf16 V_WMMA_F32_16X16X32_BF16 + GLOBAL_LOAD_ASYNC_TO_LDS double-buffering.
// ---------------------------------------------------------------------------

typedef __attribute__((ext_vector_type(16))) __bf16 v16bf;
typedef __attribute__((ext_vector_type(8)))  float  v8f;

#define DEV __device__ __forceinline__

static constexpr int Bz  = 4;
static constexpr int Sq  = 4096;
static constexpr int Dm  = 1024;
static constexpr int Hh  = 8;
static constexpr int DH  = 128;
static constexpr int WINw = 256;
static constexpr int NC  = 16;

union Frag {
    uint4          q[2];
    unsigned short u[16];
    v16bf          v;
};

DEV unsigned short f2bf(float f) {
    unsigned u = __float_as_uint(f);
    u += 0x7fffu + ((u >> 16) & 1u);          // round-to-nearest-even
    return (unsigned short)(u >> 16);
}

DEV v8f zero8() {
    v8f z = {0.f, 0.f, 0.f, 0.f, 0.f, 0.f, 0.f, 0.f};
    return z;
}

// --- CDNA5 async global->LDS DMA (ASYNCcnt) --------------------------------
// Generic pointers to __shared__ carry the LDS byte offset in their low 32
// bits; hardware adds the wave's LDS_BASE (ISA 15.18.3 async addressing).
DEV void async_ld_b128(void* lds, const void* gaddr) {
    unsigned      l = (unsigned)(uintptr_t)lds;
    unsigned long long g = (unsigned long long)(uintptr_t)gaddr;
    asm volatile("global_load_async_to_lds_b128 %0, %1, off"
                 :: "v"(l), "v"(g) : "memory");
}
template <int N>
DEV void wait_async() {
    asm volatile("s_wait_asynccnt %0" :: "i"(N) : "memory");
}

// ---------------------------------------------------------------------------
// fp32 -> bf16 elementwise (for x)
// ---------------------------------------------------------------------------
__global__ __launch_bounds__(256)
void k_f32_to_bf16(const float* __restrict__ in, unsigned short* __restrict__ out, int n4) {
    int i = blockIdx.x * blockDim.x + threadIdx.x;
    if (i < n4) {
        float4 f = ((const float4*)in)[i];
        uint2  p;
        p.x = (unsigned)f2bf(f.x) | ((unsigned)f2bf(f.y) << 16);
        p.y = (unsigned)f2bf(f.z) | ((unsigned)f2bf(f.w) << 16);
        ((uint2*)out)[i] = p;
    }
}

// ---------------------------------------------------------------------------
// W[K][N] fp32 -> Wt[N][K] bf16  (tiled transpose so GEMMs are all NT)
// ---------------------------------------------------------------------------
__global__ __launch_bounds__(256)
void k_transpose_bf16(const float* __restrict__ W, unsigned short* __restrict__ Wt,
                      int K, int N) {
    __shared__ float t[32][33];
    int tx = threadIdx.x, ty = threadIdx.y;           // (32,8)
    int n0 = blockIdx.x * 32, k0 = blockIdx.y * 32;
    #pragma unroll
    for (int i = 0; i < 4; ++i)
        t[ty + 8 * i][tx] = W[(size_t)(k0 + ty + 8 * i) * N + n0 + tx];
    __syncthreads();
    #pragma unroll
    for (int i = 0; i < 4; ++i)
        Wt[(size_t)(n0 + ty + 8 * i) * K + k0 + tx] = f2bf(t[tx][ty + 8 * i]);
}

// ---------------------------------------------------------------------------
// Generic NT GEMM: C[M][N] = A[M][K](bf16) * Bt[N][K](bf16)^T  (+ epilogue)
// Block 256 = 8 waves (4x2), tile 128x128, BK=32, LDS double-buffered with
// GLOBAL_LOAD_ASYNC_TO_LDS_B128 (copy of tile k+1 overlaps compute of tile k).
// epilogue modes:
//   0: f32  = acc + bias[n] (+ res[m*N+n])
//   1: bf16 = gelu(acc + bias[n])            (exact erf)
//   2: bf16 scatter -> dst[((b*H+h)*S+s)*DH+dd]     (q,k)
//   3: bf16 scatter -> dst[((b*H+h)*DH+dd)*S+s]     (v transposed)
// M,N multiples of 128; K multiple of 32.
// ---------------------------------------------------------------------------
__global__ __launch_bounds__(256)
void k_gemm_nt(const unsigned short* __restrict__ A,
               const unsigned short* __restrict__ Bt,
               const float* __restrict__ bias,
               const float* __restrict__ res,
               void* __restrict__ dst,
               int M, int N, int K, int mode) {
    constexpr int LDT = 40;                           // 32 + 8 pad (bf16 units)
    __shared__ unsigned short sA[2][128 * LDT];       // 2 x 10 KB
    __shared__ unsigned short sB[2][128 * LDT];       // 2 x 10 KB

    const int tid  = threadIdx.x;
    const int wave = tid >> 5, lane = tid & 31;
    const int half = lane >> 4, lm = lane & 15;
    const int wm = wave & 3, wn = wave >> 2;          // 4 x 2 wave grid
    const int m0 = blockIdx.y * 128, n0 = blockIdx.x * 128;

    // this thread's two async chunks per array (128 rows x 4 b128-chunks)
    const int r0 = tid >> 2, c0 = (tid & 3) * 8;
    const int r1 = (tid + 256) >> 2, c1 = ((tid + 256) & 3) * 8;

    auto stage = [&](int k0, int buf) {
        async_ld_b128(&sA[buf][r0 * LDT + c0], A  + (size_t)(m0 + r0) * K + k0 + c0);
        async_ld_b128(&sA[buf][r1 * LDT + c1], A  + (size_t)(m0 + r1) * K + k0 + c1);
        async_ld_b128(&sB[buf][r0 * LDT + c0], Bt + (size_t)(n0 + r0) * K + k0 + c0);
        async_ld_b128(&sB[buf][r1 * LDT + c1], Bt + (size_t)(n0 + r1) * K + k0 + c1);
    };

    v8f acc[2][4];
    #pragma unroll
    for (int tm = 0; tm < 2; ++tm)
        #pragma unroll
        for (int tn = 0; tn < 4; ++tn) acc[tm][tn] = zero8();

    const int nk = K >> 5;
    stage(0, 0);                                      // prologue: tile 0

    for (int ik = 0; ik < nk; ++ik) {
        const int ib = ik & 1;
        if (ik + 1 < nk) {
            __syncthreads();                          // readers of buf ib^1 done
            stage((ik + 1) << 5, ib ^ 1);             // async copy next tile
            if (ik + 2 < nk)                          // L2 hint for tile+2
                __builtin_prefetch(A + (size_t)(m0 + (tid >> 1)) * K + ((ik + 2) << 5), 0, 1);
            wait_async<4>();                          // tile ik landed
        } else {
            wait_async<0>();
        }
        __syncthreads();                              // all waves' tile ik in LDS

        Frag af[2], bf_[4];
        #pragma unroll
        for (int tm = 0; tm < 2; ++tm) {
            const unsigned short* p =
                &sA[ib][(32 * wm + 16 * tm + lm) * LDT + 8 * half];
            af[tm].q[0] = *(const uint4*)p;
            af[tm].q[1] = *(const uint4*)(p + 16);
        }
        #pragma unroll
        for (int tn = 0; tn < 4; ++tn) {
            const unsigned short* p =
                &sB[ib][(64 * wn + 16 * tn + lm) * LDT + 8 * half];
            bf_[tn].q[0] = *(const uint4*)p;
            bf_[tn].q[1] = *(const uint4*)(p + 16);
        }
        #pragma unroll
        for (int tm = 0; tm < 2; ++tm)
            #pragma unroll
            for (int tn = 0; tn < 4; ++tn)
                acc[tm][tn] = __builtin_amdgcn_wmma_f32_16x16x32_bf16(
                    false, af[tm].v, false, bf_[tn].v,
                    (short)0, acc[tm][tn], false, false);
    }

    // epilogue (C layout: VGPR r -> row r + 8*(lane/16); lane%16 -> col)
    #pragma unroll
    for (int tm = 0; tm < 2; ++tm)
        #pragma unroll
        for (int tn = 0; tn < 4; ++tn)
            #pragma unroll
            for (int r = 0; r < 8; ++r) {
                int m = m0 + 32 * wm + 16 * tm + r + 8 * half;
                int n = n0 + 64 * wn + 16 * tn + lm;
                float v = acc[tm][tn][r] + (bias ? bias[n] : 0.f);
                size_t mn = (size_t)m * N + n;
                if (mode == 0) {
                    ((float*)dst)[mn] = v + (res ? res[mn] : 0.f);
                } else if (mode == 1) {
                    float g = 0.5f * v * (1.f + erff(v * 0.70710678118654752f));
                    ((unsigned short*)dst)[mn] = f2bf(g);
                } else if (mode == 2) {
                    int b = m >> 12, s = m & 4095, h = n >> 7, dd = n & 127;
                    ((unsigned short*)dst)[(((size_t)(b * Hh + h)) * Sq + s) * DH + dd] = f2bf(v);
                } else {
                    int b = m >> 12, s = m & 4095, h = n >> 7, dd = n & 127;
                    ((unsigned short*)dst)[(((size_t)(b * Hh + h)) * DH + dd) * Sq + s] = f2bf(v);
                }
            }
}

// ---------------------------------------------------------------------------
// Fused sliding-window flash attention.
// grid (nc, H, B), block 512 = 16 waves; wave owns 16 query rows of its chunk.
// Streams 32-key blocks of the 3-chunk neighborhood (validity is chunk-
// uniform since 256 % 32 == 0 -> invalid neighbor chunks simply skipped).
// K/V tiles double-buffered via GLOBAL_LOAD_ASYNC_TO_LDS_B128.
// qb,kb: [B,H,S,dh] bf16;  vt: [B,H,dh,S] bf16;  ob out: [B,S,D] bf16.
// ---------------------------------------------------------------------------
__global__ __launch_bounds__(512)
void k_attn(const unsigned short* __restrict__ qb,
            const unsigned short* __restrict__ kb,
            const unsigned short* __restrict__ vt,
            unsigned short* __restrict__ ob) {
    constexpr int LDK = 136;                          // 128 + 8 pad
    constexpr int LDV = 40;                           // 32 + 8 pad
    __shared__ unsigned short sK[2][32 * LDK];        // 32 keys x 128 dh
    __shared__ unsigned short sV[2][128 * LDV];       // 128 dh x 32 keys (V^T)
    __shared__ unsigned short sP[16 * 16 * 32];       // per-wave P tiles

    const int c = blockIdx.x, h = blockIdx.y, b = blockIdx.z;
    const int tid  = threadIdx.x;
    const int wave = tid >> 5, lane = tid & 31;
    const int half = lane >> 4, lm = lane & 15;
    const size_t bh = (size_t)b * Hh + h;

    // this thread's async chunks: one K chunk, one V chunk per tile
    const int kr = tid >> 4, kc = (tid & 15) * 8;     // K: 32 rows x 16 chunks
    const int vr = tid >> 2, vc = (tid & 3) * 8;      // V: 128 rows x 4 chunks

    auto stage = [&](int j0, int buf) {
        const int gk0 = c * 256 - 256 + j0;           // global key base (valid)
        async_ld_b128(&sK[buf][kr * LDK + kc],
                      kb + (bh * Sq + gk0 + kr) * DH + kc);
        async_ld_b128(&sV[buf][vr * LDV + vc],
                      vt + (bh * DH + vr) * Sq + gk0 + vc);
    };

    // Q fragments for this wave's 16 rows, all 4 K-steps of dh=128 (kept live)
    Frag qf[4];
    {
        const unsigned short* qrow =
            qb + (bh * Sq + (size_t)c * 256 + 16 * wave + lm) * DH;
        #pragma unroll
        for (int kk = 0; kk < 4; ++kk) {
            const unsigned short* p = qrow + 32 * kk + 8 * half;
            qf[kk].q[0] = *(const uint4*)p;
            qf[kk].q[1] = *(const uint4*)(p + 16);
        }
    }

    v8f   o[8];
    float mrun[8], lrun[8];
    #pragma unroll
    for (int nt = 0; nt < 8; ++nt) o[nt] = zero8();
    #pragma unroll
    for (int r = 0; r < 8; ++r) { mrun[r] = -3.0e38f; lrun[r] = 0.f; }

    const int j0lo = (c == 0) ? 256 : 0;              // skip missing prev chunk
    const int j0hi = (c == NC - 1) ? 512 : 768;       // skip missing next chunk
    const float scale = 0.08838834764831845f;         // dh^-0.5

    stage(j0lo, 0);                                   // prologue tile
    int ib = 0;
    for (int j0 = j0lo; j0 < j0hi; j0 += 32, ib ^= 1) {
        if (j0 + 32 < j0hi) {
            __syncthreads();                          // readers of buf ib^1 done
            stage(j0 + 32, ib ^ 1);
            wait_async<2>();                          // current tile landed
        } else {
            wait_async<0>();
        }
        __syncthreads();

        // S = Q * K^T  (16 x 32 keys per wave)
        v8f sacc[2];
        #pragma unroll
        for (int tn = 0; tn < 2; ++tn) {
            sacc[tn] = zero8();
            #pragma unroll
            for (int kk = 0; kk < 4; ++kk) {
                Frag bk_;
                const unsigned short* p =
                    &sK[ib][(16 * tn + lm) * LDK + 32 * kk + 8 * half];
                bk_.q[0] = *(const uint4*)p;
                bk_.q[1] = *(const uint4*)(p + 16);
                sacc[tn] = __builtin_amdgcn_wmma_f32_16x16x32_bf16(
                    false, qf[kk].v, false, bk_.v, (short)0, sacc[tn], false, false);
            }
        }

        // band mask + scale
        float sv[2][8];
        #pragma unroll
        for (int tn = 0; tn < 2; ++tn)
            #pragma unroll
            for (int r = 0; r < 8; ++r) {
                int qi  = 16 * wave + r + 8 * half;
                int kj  = j0 + 16 * tn + lm;
                int rel = kj - WINw - qi;
                sv[tn][r] = (rel <= WINw && rel >= -WINw)
                              ? sacc[tn][r] * scale : -1e30f;
            }

        // online softmax: per-row max/sum via 16-lane shuffle reductions
        float fac[8];
        #pragma unroll
        for (int r = 0; r < 8; ++r) {
            float mloc = fmaxf(sv[0][r], sv[1][r]);
            #pragma unroll
            for (int off = 1; off < 16; off <<= 1)
                mloc = fmaxf(mloc, __shfl_xor(mloc, off, 32));
            float mnew = fmaxf(mrun[r], mloc);
            fac[r] = __expf(mrun[r] - mnew);
            mrun[r] = mnew;
            float rs = 0.f;
            #pragma unroll
            for (int tn = 0; tn < 2; ++tn) {
                float p = __expf(sv[tn][r] - mnew);
                sv[tn][r] = p;
                rs += p;
            }
            #pragma unroll
            for (int off = 1; off < 16; off <<= 1)
                rs += __shfl_xor(rs, off, 32);
            lrun[r] = lrun[r] * fac[r] + rs;
        }

        // rescale running output
        #pragma unroll
        for (int nt = 0; nt < 8; ++nt)
            #pragma unroll
            for (int r = 0; r < 8; ++r) o[nt][r] *= fac[r];

        // P through wave-private LDS: C-layout -> A-fragment layout
        unsigned short* pw = &sP[wave * (16 * 32)];
        #pragma unroll
        for (int tn = 0; tn < 2; ++tn)
            #pragma unroll
            for (int r = 0; r < 8; ++r)
                pw[(r + 8 * half) * 32 + 16 * tn + lm] = f2bf(sv[tn][r]);
        // (same-wave LDS ops are in-order; no workgroup barrier needed)

        // O += P * V   (K = 32 keys, one WMMA per dh tile)
        #pragma unroll
        for (int nt = 0; nt < 8; ++nt) {
            Frag pa, vb;
            const unsigned short* pp = pw + lm * 32 + 8 * half;
            pa.q[0] = *(const uint4*)pp;
            pa.q[1] = *(const uint4*)(pp + 16);
            const unsigned short* vp = &sV[ib][(16 * nt + lm) * LDV + 8 * half];
            vb.q[0] = *(const uint4*)vp;
            vb.q[1] = *(const uint4*)(vp + 16);
            o[nt] = __builtin_amdgcn_wmma_f32_16x16x32_bf16(
                false, pa.v, false, vb.v, (short)0, o[nt], false, false);
        }
    }

    // normalize and emit context in [B,S,D] bf16 (direct GEMM-A layout for Wo)
    #pragma unroll
    for (int r = 0; r < 8; ++r) lrun[r] = 1.f / lrun[r];
    #pragma unroll
    for (int nt = 0; nt < 8; ++nt)
        #pragma unroll
        for (int r = 0; r < 8; ++r) {
            int s = c * 256 + 16 * wave + r + 8 * half;
            int d = h * DH + 16 * nt + lm;
            ob[((size_t)b * Sq + s) * Dm + d] = f2bf(o[nt][r] * lrun[r]);
        }
}

// ---------------------------------------------------------------------------
// LayerNorm over D=1024: x2 = LN(y1)*gamma+beta (f32 + bf16 copies)
// ---------------------------------------------------------------------------
__global__ __launch_bounds__(256)
void k_ln(const float* __restrict__ y, const float* __restrict__ gamma,
          const float* __restrict__ beta, float* __restrict__ x2,
          unsigned short* __restrict__ x2b) {
    __shared__ float red[16];
    const size_t t = blockIdx.x;
    const int tid = threadIdx.x, wave = tid >> 5, lane = tid & 31;
    float4 f = ((const float4*)(y + t * Dm))[tid];
    float s  = f.x + f.y + f.z + f.w;
    float s2 = f.x * f.x + f.y * f.y + f.z * f.z + f.w * f.w;
    #pragma unroll
    for (int off = 16; off >= 1; off >>= 1) {
        s  += __shfl_xor(s,  off, 32);
        s2 += __shfl_xor(s2, off, 32);
    }
    if (lane == 0) { red[wave] = s; red[8 + wave] = s2; }
    __syncthreads();
    if (tid == 0) {
        float a = 0.f, b2 = 0.f;
        #pragma unroll
        for (int i = 0; i < 8; ++i) { a += red[i]; b2 += red[8 + i]; }
        red[0] = a; red[1] = b2;
    }
    __syncthreads();
    float mu  = red[0] * (1.f / Dm);
    float var = red[1] * (1.f / Dm) - mu * mu;
    float rs  = rsqrtf(var + 1e-5f);
    float4 g = ((const float4*)gamma)[tid];
    float4 bb = ((const float4*)beta)[tid];
    float4 o;
    o.x = (f.x - mu) * rs * g.x + bb.x;
    o.y = (f.y - mu) * rs * g.y + bb.y;
    o.z = (f.z - mu) * rs * g.z + bb.z;
    o.w = (f.w - mu) * rs * g.w + bb.w;
    ((float4*)(x2 + t * Dm))[tid] = o;
    uint2 p;
    p.x = (unsigned)f2bf(o.x) | ((unsigned)f2bf(o.y) << 16);
    p.y = (unsigned)f2bf(o.z) | ((unsigned)f2bf(o.w) << 16);
    ((uint2*)(x2b + t * Dm))[tid] = p;
}

// ---------------------------------------------------------------------------
// Host side
// ---------------------------------------------------------------------------
extern "C" void kernel_launch(void* const* d_in, const int* in_sizes, int n_in,
                              void* d_out, int out_size, void* d_ws, size_t ws_size,
                              hipStream_t stream) {
    (void)in_sizes; (void)n_in; (void)out_size; (void)ws_size;
    const float* x     = (const float*)d_in[0];
    const float* Wq    = (const float*)d_in[1];
    const float* bq    = (const float*)d_in[2];
    const float* Wk    = (const float*)d_in[3];
    const float* bk    = (const float*)d_in[4];
    const float* Wv    = (const float*)d_in[5];
    const float* bv    = (const float*)d_in[6];
    const float* Wo    = (const float*)d_in[7];
    const float* bo    = (const float*)d_in[8];
    const float* gamma = (const float*)d_in[9];
    const float* beta  = (const float*)d_in[10];
    const float* W1    = (const float*)d_in[11];
    const float* b1    = (const float*)d_in[12];
    const float* W2    = (const float*)d_in[13];
    const float* b2    = (const float*)d_in[14];

    constexpr size_t MiB = 1ull << 20;
    char* ws = (char*)d_ws;
    unsigned short* wqt = (unsigned short*)(ws + 0 * MiB);    // 2 MiB
    unsigned short* wkt = (unsigned short*)(ws + 2 * MiB);    // 2
    unsigned short* wvt = (unsigned short*)(ws + 4 * MiB);    // 2
    unsigned short* wot = (unsigned short*)(ws + 6 * MiB);    // 2
    unsigned short* w1t = (unsigned short*)(ws + 8 * MiB);    // 8
    unsigned short* w2t = (unsigned short*)(ws + 16 * MiB);   // 8
    unsigned short* xb  = (unsigned short*)(ws + 24 * MiB);   // 32
    unsigned short* qb  = (unsigned short*)(ws + 56 * MiB);   // 32
    unsigned short* kb  = (unsigned short*)(ws + 88 * MiB);   // 32
    unsigned short* vtb = (unsigned short*)(ws + 120 * MiB);  // 32
    unsigned short* obf = (unsigned short*)(ws + 152 * MiB);  // 32
    float*          y1  = (float*)(ws + 184 * MiB);           // 64
    float*          x2  = (float*)(ws + 248 * MiB);           // 64
    unsigned short* x2b = (unsigned short*)(ws + 312 * MiB);  // 32
    unsigned short* hb  = (unsigned short*)(ws + 344 * MiB);  // 128 -> 472 MiB

    const int M = Bz * Sq;                     // 16384
    const int nElem = M * Dm;                  // 16.8M

    // 1) precision conversion / weight transposes
    k_f32_to_bf16<<<(nElem / 4 + 255) / 256, 256, 0, stream>>>(x, xb, nElem / 4);
    dim3 tb(32, 8);
    k_transpose_bf16<<<dim3(Dm / 32, Dm / 32), tb, 0, stream>>>(Wq, wqt, Dm, Dm);
    k_transpose_bf16<<<dim3(Dm / 32, Dm / 32), tb, 0, stream>>>(Wk, wkt, Dm, Dm);
    k_transpose_bf16<<<dim3(Dm / 32, Dm / 32), tb, 0, stream>>>(Wv, wvt, Dm, Dm);
    k_transpose_bf16<<<dim3(Dm / 32, Dm / 32), tb, 0, stream>>>(Wo, wot, Dm, Dm);
    k_transpose_bf16<<<dim3(4096 / 32, Dm / 32), tb, 0, stream>>>(W1, w1t, Dm, 4096);
    k_transpose_bf16<<<dim3(Dm / 32, 4096 / 32), tb, 0, stream>>>(W2, w2t, 4096, Dm);

    // 2) QKV projections (scatter into attention layouts)
    dim3 gP(Dm / 128, M / 128);
    k_gemm_nt<<<gP, 256, 0, stream>>>(xb, wqt, bq, nullptr, qb,  M, Dm, Dm, 2);
    k_gemm_nt<<<gP, 256, 0, stream>>>(xb, wkt, bk, nullptr, kb,  M, Dm, Dm, 2);
    k_gemm_nt<<<gP, 256, 0, stream>>>(xb, wvt, bv, nullptr, vtb, M, Dm, Dm, 3);

    // 3) fused sliding-window flash attention
    k_attn<<<dim3(NC, Hh, Bz), 512, 0, stream>>>(qb, kb, vtb, obf);

    // 4) output projection + residual: y1 = x + o @ Wo + bo
    k_gemm_nt<<<gP, 256, 0, stream>>>(obf, wot, bo, x, y1, M, Dm, Dm, 0);

    // 5) LayerNorm
    k_ln<<<M, 256, 0, stream>>>(y1, gamma, beta, x2, x2b);

    // 6) FFN: h = gelu(x2 @ W1 + b1);  out = x2 + h @ W2 + b2
    k_gemm_nt<<<dim3(4096 / 128, M / 128), 256, 0, stream>>>(
        x2b, w1t, b1, nullptr, hb, M, 4096, Dm, 1);
    k_gemm_nt<<<gP, 256, 0, stream>>>(
        hb, w2t, b2, x2, (float*)d_out, M, Dm, 4096, 0);
}
